// GRNN_22058952032833
// MI455X (gfx1250) — compile-verified, matrix operand order checked by reading, not run on hardware
//
#include <hip/hip_runtime.h>
#include <math.h>

// ---------------------------------------------------------------------------
// Types
// ---------------------------------------------------------------------------
typedef __bf16 bhalf;
typedef bhalf bhalf16 __attribute__((ext_vector_type(16)));
typedef float float8 __attribute__((ext_vector_type(8)));

#define DEV __device__ __forceinline__

union FragU { bhalf16 v; unsigned u[8]; };

// ---------------------------------------------------------------------------
// Workspace layout (bytes). All bf16 weights stored in WMMA-fragment-swizzled
// layout: blocks of 32(K) x 16(N), 512 elems per block, element index within
// block = lane*16 + j  with  k = (p>>2)*16 + (lane>>4)*8 + ((p&3)<<1) + (j&1),
// n = lane&15, p = j>>1.  Block order: bn-major, bk-minor.
// ---------------------------------------------------------------------------
constexpr size_t GW1_B = 1048576ull * 2; // K=512 ,N=2048 swizzled bf16 bytes
constexpr size_t GW2_B = 1048576ull * 2; // K=2048,N=512
constexpr size_t SQ_B  = 262144ull * 2;  // K=512 ,N=512
constexpr size_t PL1_B = 32768ull  * 2;  // K=64  ,N=512
constexpr size_t CLS_B = 65536ull  * 2;  // K=512 ,N=128
constexpr size_t GATE_PAIR_B = GW1_B + GW2_B;

constexpr size_t OFF_GATES = 0;                              // 6 gates
constexpr size_t OFF_EDGEW = OFF_GATES + 6 * GATE_PAIR_B;    // 2 edge FFNs
constexpr size_t OFF_NODEW = OFF_EDGEW + 2 * GATE_PAIR_B;    // 2 node FFNs
constexpr size_t OFF_EPT   = OFF_NODEW + 2 * GATE_PAIR_B;    // edge proj top
constexpr size_t OFF_EPB   = OFF_EPT + SQ_B;                 // edge proj bot
constexpr size_t OFF_PL1   = OFF_EPB + SQ_B;
constexpr size_t OFF_PL2   = OFF_PL1 + PL1_B;
constexpr size_t OFF_BL1   = OFF_PL2 + SQ_B;
constexpr size_t OFF_BL2   = OFF_BL1 + PL1_B;
constexpr size_t OFF_CLS   = OFF_BL2 + SQ_B;
constexpr size_t OFF_XS    = OFF_CLS + CLS_B;                // bf16 (T,16,512)
constexpr size_t XS_BYTES  = 4096ull * 16 * 512 * 2;
constexpr size_t OFF_HF    = OFF_XS + XS_BYTES;              // f32 16x512
constexpr size_t OFF_HBF   = OFF_HF   + 32768;               // bf16 16x512
constexpr size_t OFF_TMPA  = OFF_HBF  + 16384;               // f32 16x512
constexpr size_t OFF_ACC   = OFF_TMPA + 32768;               // f32 16x512
constexpr size_t OFF_ZB    = OFF_ACC  + 32768;
constexpr size_t OFF_RB    = OFF_ZB   + 32768;
constexpr size_t OFF_HH    = OFF_RB   + 32768;
constexpr size_t OFF_RHBF  = OFF_HH   + 32768;               // bf16 16x512
constexpr size_t OFF_HID   = OFF_RHBF + 16384;               // bf16 128x2048
constexpr size_t OFF_EAF   = OFF_HID  + 128ull * 2048 * 2;   // f32 16x512
constexpr size_t OFF_EBF   = OFF_EAF  + 32768;               // f32 16x512
constexpr size_t OFF_EF    = OFF_EBF  + 32768;               // f32 128x512
constexpr size_t OFF_EBFM  = OFF_EF   + 262144;              // bf16 128x512
constexpr size_t OFF_ETF   = OFF_EBFM + 131072;              // f32 128x512
constexpr size_t OFF_OF    = OFF_ETF  + 262144;              // f32 16x512
constexpr size_t OFF_OBF   = OFF_OF   + 32768;               // bf16 16x512
constexpr size_t OFF_OTF   = OFF_OBF  + 16384;               // f32 16x512
constexpr size_t OFF_SYNC  = OFF_OTF  + 32768;               // barrier state

// ---------------------------------------------------------------------------
// WMMA fragment helpers (bf16 16x16x32, f32 accumulate)
// ---------------------------------------------------------------------------
DEV bhalf16 load_afrag(const bhalf* A, int lda, int lane) {
  // A points at (row 0, k0) of a 16x32 tile, row-major bf16
  const bhalf* ar = A + (size_t)(lane & 15) * lda;
  const int kh = (lane >> 4) << 3;
  FragU f;
#pragma unroll
  for (int p = 0; p < 8; ++p) {
    int kk = ((p >> 2) << 4) + kh + ((p & 3) << 1);
    f.u[p] = *(const unsigned*)(ar + kk);
  }
  return f.v;
}

DEV bhalf16 load_bfrag(const bhalf* Bblk, int lane) {
  // Bblk points at one 512-element swizzled 32x16 block: contiguous per lane
  FragU f;
  const unsigned* p = (const unsigned*)(Bblk + lane * 16);
#pragma unroll
  for (int i = 0; i < 8; ++i) f.u[i] = p[i];
  return f.v;
}

DEV float8 wmma_bf(bhalf16 a, bhalf16 b, float8 c) {
  return __builtin_amdgcn_wmma_f32_16x16x32_bf16(false, a, false, b, (short)0, c,
                                                 false, false);
}

// ---------------------------------------------------------------------------
// Cooperative GEMM: C[MxN] = act(A[MxK](bf16,row-major) @ Bsw[KxN](swizzled)
//                             + bias).
// Work unit = 16x64 output block: per k-step we first issue ALL loads (1 A
// fragment + 4 B fragments -> one big load clause, single wait), then 4
// back-to-back independent WMMAs (hides the bf16 WMMA->WMMA RAW hazard and
// keeps 10 b128 loads in flight per wave).  Requires N % 64 == 0.
// ---------------------------------------------------------------------------
__device__ __noinline__ void gemm_coop(const bhalf* A, int lda, const bhalf* Bsw,
                                       int M, int N, int K, const float* bias,
                                       int relu, float* Cf, bhalf* Cbf, int ldc,
                                       int mstore, int wave, int nwaves) {
  const int ng = N >> 6;                 // groups of 4 n-tiles
  const int kb = K >> 5;
  const int total = (M >> 4) * ng;
  const int lane = threadIdx.x & 31;
  for (int grp = wave; grp < total; grp += nwaves) {
    const int mt = grp / ng;
    const int m0 = mt << 4;
    const int n0 = (grp - mt * ng) << 6;
    float8 acc[4];
#pragma unroll
    for (int i = 0; i < 4; ++i) acc[i] = float8{0.f, 0.f, 0.f, 0.f, 0.f, 0.f, 0.f, 0.f};
    const bhalf* abase = A + (size_t)m0 * lda;
    const bhalf* bbase = Bsw + (size_t)(n0 >> 4) * kb * 512;
    for (int k = 0; k < kb; ++k) {
      if (k + 1 < kb)
        __builtin_prefetch(bbase + (size_t)(k + 1) * 512, 0, 0);
      // issue all loads first -> one clause, one wait, then 4 indep. WMMAs
      bhalf16 bf[4];
#pragma unroll
      for (int i = 0; i < 4; ++i)
        bf[i] = load_bfrag(bbase + ((size_t)i * kb + k) * 512, lane);
      bhalf16 af = load_afrag(abase + (k << 5), lda, lane);
#pragma unroll
      for (int i = 0; i < 4; ++i) acc[i] = wmma_bf(af, bf[i], acc[i]);
    }
    const int mb = m0 + ((lane >> 4) << 3);
#pragma unroll
    for (int i = 0; i < 4; ++i) {
      const int n = n0 + (i << 4) + (lane & 15);
      const float bn = bias ? bias[n] : 0.f;
#pragma unroll
      for (int r = 0; r < 8; ++r) {
        int m = mb + r;
        if (m >= mstore) continue;
        float val = acc[i][r] + bn;
        if (relu) val = fmaxf(val, 0.f);
        if (Cf) Cf[(size_t)m * ldc + n] = val;
        if (Cbf) Cbf[(size_t)m * ldc + n] = (bhalf)val;
      }
    }
  }
}

// ---------------------------------------------------------------------------
// Cooperative LayerNorm over 512-wide rows (one row per wave).
//   y = g * LN(res + F) + b ;  act: 0 = store y
//                              1 = sigmoid(y + acc + gbias)
//                              2 = tanh   (y + acc + gbias)
//   optional: mulOut = bf16(out * mulH) (used for r*h)
// ---------------------------------------------------------------------------
__device__ __noinline__ void ln_stage(const float* resF, const bhalf* resBf,
                                      const float* F, const float* g,
                                      const float* b, const float* acc,
                                      const float* gbias, int act, float* outF,
                                      bhalf* outBf, const float* mulH,
                                      bhalf* mulOut, int rows, int wave,
                                      int nwaves) {
  const int lane = threadIdx.x & 31;
  for (int row = wave; row < rows; row += nwaves) {
    const int base = row * 512;
    float v[16];
    float s = 0.f;
#pragma unroll
    for (int i = 0; i < 16; ++i) {
      int c = lane + (i << 5);
      float xr = resF ? resF[base + c] : (float)resBf[base + c];
      v[i] = xr + F[base + c];
      s += v[i];
    }
#pragma unroll
    for (int o = 16; o > 0; o >>= 1) s += __shfl_xor(s, o, 32);
    const float mu = s * (1.f / 512.f);
    float vs = 0.f;
#pragma unroll
    for (int i = 0; i < 16; ++i) {
      v[i] -= mu;
      vs += v[i] * v[i];
    }
#pragma unroll
    for (int o = 16; o > 0; o >>= 1) vs += __shfl_xor(vs, o, 32);
    const float rs = rsqrtf(vs * (1.f / 512.f) + 1e-5f);
#pragma unroll
    for (int i = 0; i < 16; ++i) {
      int c = lane + (i << 5);
      float y = g[c] * (v[i] * rs) + b[c];
      if (act) {
        y += acc[base + c] + gbias[c];
        y = (act == 1) ? 1.f / (1.f + __expf(-y)) : tanhf(y);
      }
      if (outF) outF[base + c] = y;
      if (outBf) outBf[base + c] = (bhalf)y;
      if (mulOut) mulOut[base + c] = (bhalf)(y * mulH[base + c]);
    }
  }
}

// ---------------------------------------------------------------------------
// Device-wide sense-reversal barrier (all WGs co-resident: 64 x 256 threads)
// ---------------------------------------------------------------------------
DEV void grid_barrier(unsigned* cnt, unsigned* gen, unsigned nwg) {
  __threadfence();
  __syncthreads();
  if (threadIdx.x == 0) {
    unsigned g = __hip_atomic_load(gen, __ATOMIC_RELAXED, __HIP_MEMORY_SCOPE_AGENT);
    unsigned a =
        __hip_atomic_fetch_add(cnt, 1u, __ATOMIC_ACQ_REL, __HIP_MEMORY_SCOPE_AGENT);
    if (a == nwg - 1u) {
      __hip_atomic_store(cnt, 0u, __ATOMIC_RELAXED, __HIP_MEMORY_SCOPE_AGENT);
      __hip_atomic_fetch_add(gen, 1u, __ATOMIC_ACQ_REL, __HIP_MEMORY_SCOPE_AGENT);
    } else {
      unsigned cur;
      do {
        __builtin_amdgcn_s_sleep(1);
        cur = __hip_atomic_load(gen, __ATOMIC_ACQUIRE, __HIP_MEMORY_SCOPE_AGENT);
      } while (cur == g);
    }
  }
  __syncthreads();
  __threadfence();
}

// ---------------------------------------------------------------------------
// Weight conversion: fp32 [K x N] row-major -> bf16 swizzled blocks.
// ---------------------------------------------------------------------------
__global__ __launch_bounds__(256) void conv_swizzle_kernel(const float* src,
                                                           bhalf* dst, int Ksrc,
                                                           int Kpad, int N,
                                                           int total) {
  for (int e = blockIdx.x * 256 + threadIdx.x; e < total;
       e += gridDim.x * 256) {
    int block = e >> 9;
    int w = e & 511;
    int lane = w >> 4;
    int j = w & 15;
    int kbn = Kpad >> 5;
    int bk = block % kbn;
    int bn = block / kbn;
    int p = j >> 1;
    int k = (bk << 5) + ((p >> 2) << 4) + ((lane >> 4) << 3) + ((p & 3) << 1) +
            (j & 1);
    int n = (bn << 4) + (lane & 15);
    dst[e] = (k < Ksrc) ? (bhalf)src[(size_t)k * N + n] : (bhalf)0.f;
  }
}

// ---------------------------------------------------------------------------
// h0 init: rows 0..10 from params, rows 11..15 zero (f32 + bf16 mirror)
// ---------------------------------------------------------------------------
__global__ __launch_bounds__(256) void hinit_kernel(const float* h0, float* hF,
                                                    bhalf* hBf) {
  int i = blockIdx.x * 256 + threadIdx.x;
  if (i < 8192) {
    int r = i >> 9, c = i & 511;
    float v = (r < 11) ? h0[r * 512 + c] : 0.f;
    hF[i] = v;
    hBf[i] = (bhalf)v;
  }
}

// ---------------------------------------------------------------------------
// Featurization (parallel over T):   feats(35,padK=64) -> 512 relu -> 512
// Each of the 8 waves owns one 16x64 output group per GEMM (4 accumulators).
// ---------------------------------------------------------------------------
DEV void feat_gemm1(const bhalf* Ain, const bhalf* w1sw, const float* b1,
                    bhalf* Hid, int wave, int lane) {
  // M=16, N=512, K=64 (kb=2): Hid = relu(Ain @ W1 + b1), bf16 into LDS
  for (int grp = wave; grp < 8; grp += 8) {
    int n0 = grp << 6;
    float8 acc[4];
#pragma unroll
    for (int i = 0; i < 4; ++i) acc[i] = float8{0.f, 0.f, 0.f, 0.f, 0.f, 0.f, 0.f, 0.f};
    const bhalf* bbase = w1sw + (size_t)(n0 >> 4) * 2 * 512;
#pragma unroll
    for (int k = 0; k < 2; ++k) {
      bhalf16 bf[4];
#pragma unroll
      for (int i = 0; i < 4; ++i)
        bf[i] = load_bfrag(bbase + ((size_t)i * 2 + k) * 512, lane);
      bhalf16 af = load_afrag(Ain + (k << 5), 64, lane);
#pragma unroll
      for (int i = 0; i < 4; ++i) acc[i] = wmma_bf(af, bf[i], acc[i]);
    }
    int mb = (lane >> 4) << 3;
#pragma unroll
    for (int i = 0; i < 4; ++i) {
      int n = n0 + (i << 4) + (lane & 15);
#pragma unroll
      for (int r = 0; r < 8; ++r)
        Hid[(mb + r) * 512 + n] = (bhalf)fmaxf(acc[i][r] + b1[n], 0.f);
    }
  }
}

__global__ __launch_bounds__(256) void feat_players_kernel(
    const int* idx, const float* pxs, const float* pys, const float* phs,
    const float* emb, const bhalf* w1sw, const float* b1, const bhalf* w2sw,
    const float* b2, bhalf* xsout) {
  __shared__ __align__(16) bhalf Ain[16 * 64];
  __shared__ __align__(16) bhalf Hid[16 * 512];
  const int r0 = blockIdx.x * 16;  // 16 rows of (t,p) flat index
  for (int i = threadIdx.x; i < 1024; i += 256) {
    int row = i >> 6, k = i & 63;
    int gr = r0 + row;
    float v = 0.f;
    if (k < 32)
      v = emb[(size_t)idx[gr] * 32 + k];
    else if (k == 32)
      v = pxs[gr];
    else if (k == 33)
      v = pys[gr];
    else if (k == 34)
      v = phs[gr];
    Ain[i] = (bhalf)v;
  }
  __syncthreads();
  const int wave = threadIdx.x >> 5, lane = threadIdx.x & 31;
  feat_gemm1(Ain, w1sw, b1, Hid, wave, lane);
  __syncthreads();
  for (int grp = wave; grp < 8; grp += 8) {
    int n0 = grp << 6;
    float8 acc[4];
#pragma unroll
    for (int i = 0; i < 4; ++i) acc[i] = float8{0.f, 0.f, 0.f, 0.f, 0.f, 0.f, 0.f, 0.f};
    const bhalf* bbase = w2sw + (size_t)(n0 >> 4) * 16 * 512;
    for (int k = 0; k < 16; ++k) {
      bhalf16 bf[4];
#pragma unroll
      for (int i = 0; i < 4; ++i)
        bf[i] = load_bfrag(bbase + ((size_t)i * 16 + k) * 512, lane);
      bhalf16 af = load_afrag(Hid + (k << 5), 512, lane);
#pragma unroll
      for (int i = 0; i < 4; ++i) acc[i] = wmma_bf(af, bf[i], acc[i]);
    }
    int mb = (lane >> 4) << 3;
#pragma unroll
    for (int i = 0; i < 4; ++i) {
      int n = n0 + (i << 4) + (lane & 15);
#pragma unroll
      for (int r = 0; r < 8; ++r) {
        int gr = r0 + mb + r;
        int t = gr / 10, p = gr - t * 10;
        xsout[((size_t)t * 16 + p) * 512 + n] = (bhalf)(acc[i][r] + b2[n]);
      }
    }
  }
}

__global__ __launch_bounds__(256) void feat_ball_kernel(
    const float* bx, const float* by, const float* bz, const float* bemb,
    const bhalf* w1sw, const float* b1, const bhalf* w2sw, const float* b2,
    bhalf* xsout) {
  __shared__ __align__(16) bhalf Ain[16 * 64];
  __shared__ __align__(16) bhalf Hid[16 * 512];
  const int r0 = blockIdx.x * 16;  // 16 timesteps
  for (int i = threadIdx.x; i < 1024; i += 256) {
    int row = i >> 6, k = i & 63;
    int gr = r0 + row;
    float v = 0.f;
    if (k < 32)
      v = bemb[k];
    else if (k == 32)
      v = bx[gr];
    else if (k == 33)
      v = by[gr];
    else if (k == 34)
      v = bz[gr];
    Ain[i] = (bhalf)v;
  }
  __syncthreads();
  const int wave = threadIdx.x >> 5, lane = threadIdx.x & 31;
  feat_gemm1(Ain, w1sw, b1, Hid, wave, lane);
  __syncthreads();
  const float SCALE = 22.627416997969522f;  // sqrt(512)
  for (int grp = wave; grp < 8; grp += 8) {
    int n0 = grp << 6;
    float8 acc[4];
#pragma unroll
    for (int i = 0; i < 4; ++i) acc[i] = float8{0.f, 0.f, 0.f, 0.f, 0.f, 0.f, 0.f, 0.f};
    const bhalf* bbase = w2sw + (size_t)(n0 >> 4) * 16 * 512;
    for (int k = 0; k < 16; ++k) {
      bhalf16 bf[4];
#pragma unroll
      for (int i = 0; i < 4; ++i)
        bf[i] = load_bfrag(bbase + ((size_t)i * 16 + k) * 512, lane);
      bhalf16 af = load_afrag(Hid + (k << 5), 512, lane);
#pragma unroll
      for (int i = 0; i < 4; ++i) acc[i] = wmma_bf(af, bf[i], acc[i]);
    }
    int mb = (lane >> 4) << 3;
#pragma unroll
    for (int i = 0; i < 4; ++i) {
      int n = n0 + (i << 4) + (lane & 15);
#pragma unroll
      for (int r = 0; r < 8; ++r) {
        int t = r0 + mb + r;
        xsout[((size_t)t * 16 + 10) * 512 + n] =
            (bhalf)((acc[i][r] + b2[n]) * SCALE);
      }
    }
  }
}

// ---------------------------------------------------------------------------
// Persistent scan kernel
// ---------------------------------------------------------------------------
struct ScanArgs {
  const bhalf* xs;
  float* hF;
  bhalf* hBf;
  const bhalf* gw1[6];
  const bhalf* gw2[6];
  const float* gb1[6];
  const float* gb2[6];
  const float* glg[6];
  const float* glb[6];
  const float* bz;
  const float* br;
  const float* bh;
  const bhalf* ept;
  const bhalf* epb;
  const float* epbias;
  const bhalf* ew1[2];
  const bhalf* ew2[2];
  const float* eb1[2];
  const float* eb2[2];
  const float* eg[2];
  const float* ebt[2];
  const bhalf* nw1[2];
  const bhalf* nw2[2];
  const float* nb1[2];
  const float* nb2[2];
  const float* ng[2];
  const float* nbt[2];
  const bhalf* clsw;
  const float* clsb;
  float* tmpA;
  float* acc;
  float* zb;
  float* rb;
  float* hh;
  bhalf* rhBf;
  bhalf* hid;
  float* eaF;
  float* ebF;
  float* eF;
  bhalf* eBf;
  float* etF;
  float* oF;
  bhalf* oBf;
  float* otF;
  float* out;
  unsigned* cnt;
  unsigned* gen;
};

__global__ __launch_bounds__(256, 1) void scan_kernel(ScanArgs A) {
  const int wave = (blockIdx.x << 3) + (threadIdx.x >> 5);
  const int nw = gridDim.x << 3;
  const int gtid = blockIdx.x * 256 + threadIdx.x;
  const int nthr = gridDim.x * 256;
  const unsigned nwg = gridDim.x;
#define GB() grid_barrier(A.cnt, A.gen, nwg)

#pragma clang loop unroll(disable)
  for (int t = 0; t < 4096; ++t) {
    const bhalf* x = A.xs + (size_t)t * 8192;

    // ---- GRU gates: z, r, h_hat -------------------------------------
#pragma clang loop unroll(disable)
    for (int gi = 0; gi < 3; ++gi) {
      const int gw = gi * 2, gu = gi * 2 + 1;
      const float* gbias = (gi == 0) ? A.bz : (gi == 1) ? A.br : A.bh;
      const bhalf* uin = (gi == 2) ? A.rhBf : A.hBf;
      // W-side FFN(x): 16x2048x512 relu -> 16x512x2048
      gemm_coop(x, 512, A.gw1[gw], 16, 2048, 512, A.gb1[gw], 1, nullptr, A.hid,
                2048, 16, wave, nw);
      GB();
      gemm_coop(A.hid, 2048, A.gw2[gw], 16, 512, 2048, A.gb2[gw], 0, A.tmpA,
                nullptr, 512, 16, wave, nw);
      GB();
      // acc = LN(x + F)
      ln_stage(nullptr, x, A.tmpA, A.glg[gw], A.glb[gw], nullptr, nullptr, 0,
               A.acc, nullptr, nullptr, nullptr, 16, wave, nw);
      GB();
      // U-side FFN(h or r*h)
      gemm_coop(uin, 512, A.gw1[gu], 16, 2048, 512, A.gb1[gu], 1, nullptr,
                A.hid, 2048, 16, wave, nw);
      GB();
      gemm_coop(A.hid, 2048, A.gw2[gu], 16, 512, 2048, A.gb2[gu], 0, A.tmpA,
                nullptr, 512, 16, wave, nw);
      GB();
      // gate = act(LN(h + F) + acc + b); r-gate also emits rh = bf16(r*h)
      float* outF = (gi == 0) ? A.zb : (gi == 1) ? A.rb : A.hh;
      ln_stage(A.hF, nullptr, A.tmpA, A.glg[gu], A.glb[gu], A.acc, gbias,
               (gi == 2) ? 2 : 1, outF, nullptr, (gi == 1) ? A.hF : nullptr,
               (gi == 1) ? A.rhBf : nullptr, 16, wave, nw);
      GB();
    }

    // ---- h = (1-z)*h + z*h_hat --------------------------------------
    for (int i = gtid; i < 8192; i += nthr) {
      float z = A.zb[i];
      float v = (1.f - z) * A.hF[i] + z * A.hh[i];
      A.hF[i] = v;
      A.hBf[i] = (bhalf)v;
    }
    GB();

    // ---- edge projection: EA = h@Wtop, EB = h@Wbot ------------------
    gemm_coop(A.hBf, 512, A.ept, 16, 512, 512, nullptr, 0, A.eaF, nullptr, 512,
              16, wave, nw);
    gemm_coop(A.hBf, 512, A.epb, 16, 512, 512, nullptr, 0, A.ebF, nullptr, 512,
              16, wave, nw);
    GB();

    // ---- build E[q=i*11+j] = EA[i] + EB[j] + b (pad rows zero) ------
    for (int i = gtid; i < 65536; i += nthr) {
      int q = i >> 9, c = i & 511;
      float v = 0.f;
      if (q < 121) {
        int ie = q / 11, je = q - ie * 11;
        v = A.eaF[ie * 512 + c] + A.ebF[je * 512 + c] + A.epbias[c];
      }
      A.eF[i] = v;
      A.eBf[i] = (bhalf)v;
    }
    GB();

    // ---- edge GNN layers: E = LN(E + FFN(E)) ------------------------
#pragma clang loop unroll(disable)
    for (int l = 0; l < 2; ++l) {
      gemm_coop(A.eBf, 512, A.ew1[l], 128, 2048, 512, A.eb1[l], 1, nullptr,
                A.hid, 2048, 128, wave, nw);
      GB();
      gemm_coop(A.hid, 2048, A.ew2[l], 128, 512, 2048, A.eb2[l], 0, A.etF,
                nullptr, 512, 128, wave, nw);
      GB();
      ln_stage(A.eF, nullptr, A.etF, A.eg[l], A.ebt[l], nullptr, nullptr, 0,
               A.eF, A.eBf, nullptr, nullptr, 128, wave, nw);
      GB();
    }

    // ---- aggregate o[i] = sum_j e[i,j] ------------------------------
    for (int i = gtid; i < 8192; i += nthr) {
      int ie = i >> 9, c = i & 511;
      float s = 0.f;
      if (ie < 11)
        for (int j = 0; j < 11; ++j) s += A.eF[(ie * 11 + j) * 512 + c];
      A.oF[i] = s;
      A.oBf[i] = (bhalf)s;
    }
    GB();

    // ---- node layers: o = LN(o + FFN(o)) ----------------------------
#pragma clang loop unroll(disable)
    for (int l = 0; l < 2; ++l) {
      gemm_coop(A.oBf, 512, A.nw1[l], 16, 2048, 512, A.nb1[l], 1, nullptr,
                A.hid, 2048, 16, wave, nw);
      GB();
      gemm_coop(A.hid, 2048, A.nw2[l], 16, 512, 2048, A.nb2[l], 0, A.otF,
                nullptr, 512, 16, wave, nw);
      GB();
      ln_stage(A.oF, nullptr, A.otF, A.ng[l], A.nbt[l], nullptr, nullptr, 0,
               A.oF, A.oBf, nullptr, nullptr, 16, wave, nw);
      GB();
    }

    // ---- classifier: pred = o[:10] @ Wcls + b -> d_out --------------
    gemm_coop(A.oBf, 512, A.clsw, 16, 128, 512, A.clsb, 0,
              A.out + (size_t)t * 1280, nullptr, 128, 10, wave, nw);
    GB();
  }
#undef GB
}

// ---------------------------------------------------------------------------
// Host entry.  Param leaf index table (setup_inputs/make_params insertion
// order):
//  0..6 runtime inputs; 7 emb; 8 ball_emb; 9..12 player_mlp(w1,b1,w2,b2);
//  13..16 ball_mlp; 17 h0; 18 b_z; 19..24 W_z(w1,b1,w2,b2,g,b); 25..30 U_z;
//  31 b_r; 32..37 W_r; 38..43 U_r; 44 b_h; 45..50 W_h; 51..56 U_h;
//  57 edge_w; 58 edge_b; 59..66 edge_layers; 67..70 edge_norms;
//  71..78 node_layers; 79..82 node_norms; 83 cls_w; 84 cls_b.
// ---------------------------------------------------------------------------
extern "C" void kernel_launch(void* const* d_in, const int* in_sizes, int n_in,
                              void* d_out, int out_size, void* d_ws,
                              size_t ws_size, hipStream_t stream) {
  (void)in_sizes;
  (void)n_in;
  (void)out_size;
  (void)ws_size;
  char* ws = (char*)d_ws;
  auto fin = [&](int i) { return (const float*)d_in[i]; };

  auto conv = [&](const float* src, size_t dstOff, int Ksrc, int Kpad, int N) {
    int total = Kpad * N;
    conv_swizzle_kernel<<<dim3(total / 256), dim3(256), 0, stream>>>(
        src, (bhalf*)(ws + dstOff), Ksrc, Kpad, N, total);
  };

  const int w1i[6] = {19, 25, 32, 38, 45, 51};
  const int b1i[6] = {20, 26, 33, 39, 46, 52};
  const int w2i[6] = {21, 27, 34, 40, 47, 53};
  const int b2i[6] = {22, 28, 35, 41, 48, 54};
  const int lgi[6] = {23, 29, 36, 42, 49, 55};
  const int lbi[6] = {24, 30, 37, 43, 50, 56};

  for (int g = 0; g < 6; ++g) {
    conv(fin(w1i[g]), OFF_GATES + g * GATE_PAIR_B, 512, 512, 2048);
    conv(fin(w2i[g]), OFF_GATES + g * GATE_PAIR_B + GW1_B, 2048, 2048, 512);
  }
  for (int l = 0; l < 2; ++l) {
    conv(fin(59 + 4 * l), OFF_EDGEW + l * GATE_PAIR_B, 512, 512, 2048);
    conv(fin(61 + 4 * l), OFF_EDGEW + l * GATE_PAIR_B + GW1_B, 2048, 2048, 512);
    conv(fin(71 + 4 * l), OFF_NODEW + l * GATE_PAIR_B, 512, 512, 2048);
    conv(fin(73 + 4 * l), OFF_NODEW + l * GATE_PAIR_B + GW1_B, 2048, 2048, 512);
  }
  conv(fin(57), OFF_EPT, 512, 512, 512);
  conv(fin(57) + 512 * 512, OFF_EPB, 512, 512, 512);
  conv(fin(9), OFF_PL1, 35, 64, 512);
  conv(fin(11), OFF_PL2, 512, 512, 512);
  conv(fin(13), OFF_BL1, 35, 64, 512);
  conv(fin(15), OFF_BL2, 512, 512, 512);
  conv(fin(83), OFF_CLS, 512, 512, 128);

  hipMemsetAsync(ws + OFF_XS, 0, XS_BYTES, stream);
  hipMemsetAsync(ws + OFF_SYNC, 0, 256, stream);

  hinit_kernel<<<dim3(32), dim3(256), 0, stream>>>(
      fin(17), (float*)(ws + OFF_HF), (bhalf*)(ws + OFF_HBF));

  feat_players_kernel<<<dim3(2560), dim3(256), 0, stream>>>(
      (const int*)d_in[0], fin(1), fin(2), fin(3), fin(7),
      (const bhalf*)(ws + OFF_PL1), fin(10), (const bhalf*)(ws + OFF_PL2),
      fin(12), (bhalf*)(ws + OFF_XS));
  feat_ball_kernel<<<dim3(256), dim3(256), 0, stream>>>(
      fin(4), fin(5), fin(6), fin(8), (const bhalf*)(ws + OFF_BL1), fin(14),
      (const bhalf*)(ws + OFF_BL2), fin(16), (bhalf*)(ws + OFF_XS));

  ScanArgs sa{};
  sa.xs = (const bhalf*)(ws + OFF_XS);
  sa.hF = (float*)(ws + OFF_HF);
  sa.hBf = (bhalf*)(ws + OFF_HBF);
  for (int g = 0; g < 6; ++g) {
    sa.gw1[g] = (const bhalf*)(ws + OFF_GATES + g * GATE_PAIR_B);
    sa.gw2[g] = (const bhalf*)(ws + OFF_GATES + g * GATE_PAIR_B + GW1_B);
    sa.gb1[g] = fin(b1i[g]);
    sa.gb2[g] = fin(b2i[g]);
    sa.glg[g] = fin(lgi[g]);
    sa.glb[g] = fin(lbi[g]);
  }
  sa.bz = fin(18);
  sa.br = fin(31);
  sa.bh = fin(44);
  sa.ept = (const bhalf*)(ws + OFF_EPT);
  sa.epb = (const bhalf*)(ws + OFF_EPB);
  sa.epbias = fin(58);
  for (int l = 0; l < 2; ++l) {
    sa.ew1[l] = (const bhalf*)(ws + OFF_EDGEW + l * GATE_PAIR_B);
    sa.ew2[l] = (const bhalf*)(ws + OFF_EDGEW + l * GATE_PAIR_B + GW1_B);
    sa.eb1[l] = fin(60 + 4 * l);
    sa.eb2[l] = fin(62 + 4 * l);
    sa.eg[l] = fin(67 + 2 * l);
    sa.ebt[l] = fin(68 + 2 * l);
    sa.nw1[l] = (const bhalf*)(ws + OFF_NODEW + l * GATE_PAIR_B);
    sa.nw2[l] = (const bhalf*)(ws + OFF_NODEW + l * GATE_PAIR_B + GW1_B);
    sa.nb1[l] = fin(72 + 4 * l);
    sa.nb2[l] = fin(74 + 4 * l);
    sa.ng[l] = fin(79 + 2 * l);
    sa.nbt[l] = fin(80 + 2 * l);
  }
  sa.clsw = (const bhalf*)(ws + OFF_CLS);
  sa.clsb = fin(84);
  sa.tmpA = (float*)(ws + OFF_TMPA);
  sa.acc = (float*)(ws + OFF_ACC);
  sa.zb = (float*)(ws + OFF_ZB);
  sa.rb = (float*)(ws + OFF_RB);
  sa.hh = (float*)(ws + OFF_HH);
  sa.rhBf = (bhalf*)(ws + OFF_RHBF);
  sa.hid = (bhalf*)(ws + OFF_HID);
  sa.eaF = (float*)(ws + OFF_EAF);
  sa.ebF = (float*)(ws + OFF_EBF);
  sa.eF = (float*)(ws + OFF_EF);
  sa.eBf = (bhalf*)(ws + OFF_EBFM);
  sa.etF = (float*)(ws + OFF_ETF);
  sa.oF = (float*)(ws + OFF_OF);
  sa.oBf = (bhalf*)(ws + OFF_OBF);
  sa.otF = (float*)(ws + OFF_OTF);
  sa.out = (float*)d_out;
  sa.cnt = (unsigned*)(ws + OFF_SYNC);
  sa.gen = sa.cnt + 1;

  scan_kernel<<<dim3(64), dim3(256), 0, stream>>>(sa);
}
